// SpecializedGNN_57148834840787
// MI455X (gfx1250) — matrix operand order repaired and checked by reading, not compiled, not run on hardware
//
#include <hip/hip_runtime.h>
#include <hip/hip_bf16.h>

typedef __attribute__((ext_vector_type(16))) __bf16 v16bf;
typedef __attribute__((ext_vector_type(8)))  __bf16 v8bf;
typedef __attribute__((ext_vector_type(8)))  float  v8f;

#define F_IN   1024
#define F_HID  512
#define GEMM_M 32     // rows per block (two 16-row WMMA sub-tiles)

// ---------------------------------------------------------------- utilities
__global__ void gnn_fill_f32(float* p, float v, int n) {
    int i = blockIdx.x * blockDim.x + threadIdx.x;
    if (i < n) p[i] = v;
}

__global__ void gnn_deg_accum(const int* __restrict__ tgt, float* deg, int n_edges) {
    int e = blockIdx.x * blockDim.x + threadIdx.x;
    if (e < n_edges) atomicAdd(&deg[tgt[e]], 1.0f);
}

__global__ void gnn_rsqrt_inplace(float* p, int n) {
    int i = blockIdx.x * blockDim.x + threadIdx.x;
    if (i < n) p[i] = rsqrtf(p[i]);
}

// W1 [F_IN, F_HID] f32 row-major  ->  w1t [F_HID, F_IN] bf16 (transposed: K contiguous)
__global__ void gnn_pack_w1(const float* __restrict__ W1, __bf16* __restrict__ w1t) {
    int idx = blockIdx.x * blockDim.x + threadIdx.x;
    if (idx >= F_IN * F_HID) return;
    int k = idx / F_HID, n = idx % F_HID;
    w1t[(size_t)n * F_IN + k] = (__bf16)W1[idx];
}

// ---------------------------------------------------------------- GEMM: h0 = x @ W1
// One block (8 waves) computes rows [m0, m0+32) x all 512 cols.
// A tile (32x32) staged once per K-step into LDS as bf16 in WMMA fragment order;
// wave w owns N-tiles w*4..w*4+3; each wave issues 8 WMMAs per K-step.
__global__ __launch_bounds__(256) void gnn_gemm1_wmma(
    const float* __restrict__ x,       // [n, F_IN]
    const __bf16* __restrict__ w1t,    // [F_HID, F_IN] bf16
    float* __restrict__ h0,            // [n, F_HID]
    int n_nodes)
{
    // two 16x32 bf16 sub-tiles, lane-fragment ordered: [sub*512 + lane*16 + elem]
    __shared__ __align__(32) __bf16 atile[2 * 16 * 32];

    const int wave  = threadIdx.x >> 5;
    const int lane  = threadIdx.x & 31;
    const int m0    = blockIdx.x * GEMM_M;
    const int lrow  = lane & 15;
    const int lhalf = lane >> 4;

    v8f acc[2][4];
#pragma unroll
    for (int s = 0; s < 2; ++s)
#pragma unroll
        for (int t = 0; t < 4; ++t)
            acc[s][t] = (v8f){0.f,0.f,0.f,0.f,0.f,0.f,0.f,0.f};

    for (int k0 = 0; k0 < F_IN; k0 += 32) {
        __syncthreads();   // previous iteration's LDS reads complete (WAR)
        // ---- stage 32x32 f32 -> bf16 into LDS in A-fragment order.
        // A-fragment (16x32): lane = r%16 + 16*((c>>3)&1), elem = (c&7) + 8*(c>>4)
#pragma unroll
        for (int i = 0; i < 2; ++i) {
            int p = threadIdx.x * 2 + i;       // 512 col-pairs
            int r = p >> 4;                    // 0..31
            int c = (p & 15) * 2;              // 0..30 even
            int rload = m0 + r;
            if (rload >= n_nodes) rload = n_nodes - 1;
            const float2 v = *(const float2*)(x + (size_t)rload * F_IN + k0 + c);
            int sub = r >> 4;
            int ln  = (r & 15) + (((c >> 3) & 1) << 4);
            int e   = (c & 7) + ((c >> 4) << 3);
            union { unsigned u; __bf16 h[2]; } pk;
            pk.h[0] = (__bf16)v.x;
            pk.h[1] = (__bf16)v.y;
            *(unsigned*)&atile[sub * 512 + ln * 16 + e] = pk.u;
        }
        __syncthreads();

        v16bf a0 = *(const v16bf*)&atile[lane * 16];
        v16bf a1 = *(const v16bf*)&atile[512 + lane * 16];

#pragma unroll
        for (int t = 0; t < 4; ++t) {
            // ---- B fragment (32x16 bf16): lane holds col (n0+lrow),
            //      elems e<8 -> K = k0 + lhalf*8 + e, e>=8 -> K = k0 + 16 + lhalf*8 + (e-8)
            int ncol = (wave * 4 + t) * 16 + lrow;
            const __bf16* wrow = w1t + (size_t)ncol * F_IN + k0 + (lhalf << 3);
            v8bf blo = *(const v8bf*)(wrow);
            v8bf bhi = *(const v8bf*)(wrow + 16);
            v16bf b;
#pragma unroll
            for (int e = 0; e < 8; ++e) { b[e] = blo[e]; b[e + 8] = bhi[e]; }

            acc[0][t] = __builtin_amdgcn_wmma_f32_16x16x32_bf16(
                false, a0, false, b, (short)0, acc[0][t], false, false);
            acc[1][t] = __builtin_amdgcn_wmma_f32_16x16x32_bf16(
                false, a1, false, b, (short)0, acc[1][t], false, false);
        }
    }

    // ---- store C/D: lane -> N = n0 + lrow ; VGPR v -> M = msub + 8*lhalf + v
#pragma unroll
    for (int s = 0; s < 2; ++s) {
        int mbase = m0 + s * 16 + 8 * lhalf;
#pragma unroll
        for (int t = 0; t < 4; ++t) {
            int n0 = (wave * 4 + t) * 16;
#pragma unroll
            for (int v = 0; v < 8; ++v) {
                int m = mbase + v;
                if (m < n_nodes) h0[(size_t)m * F_HID + n0 + lrow] = acc[s][t][v];
            }
        }
    }
}

// ---------------------------------------------------------------- aggregation 1
// agg[i,:] = h0[i,:] * dinv[i]^2   (self-loop term)
__global__ void gnn_self_term(const float* __restrict__ h0, const float* __restrict__ dinv,
                              float* __restrict__ agg, int total) {
    int i = blockIdx.x * blockDim.x + threadIdx.x;
    if (i < total) {
        float d = dinv[i / F_HID];
        agg[i] = h0[i] * d * d;
    }
}

// one wave per edge: agg[tgt,:] += dinv[src]*dinv[tgt] * h0[src,:]
__global__ __launch_bounds__(256) void gnn_scatter_rows(
    const int* __restrict__ src, const int* __restrict__ tgt,
    const float* __restrict__ dinv, const float* __restrict__ h0,
    float* __restrict__ agg, int n_edges)
{
    int e = blockIdx.x * 8 + (threadIdx.x >> 5);
    int lane = threadIdx.x & 31;
    if (e >= n_edges) return;
    int s = src[e], t = tgt[e];
    float norm = dinv[s] * dinv[t];
    const float4* hs = (const float4*)(h0 + (size_t)s * F_HID);
    float* at = agg + (size_t)t * F_HID;
#pragma unroll
    for (int i = 0; i < 4; ++i) {
        int v4 = lane + i * 32;          // 128 float4 = 512 floats
        float4 v = hs[v4];
        int base = v4 * 4;
        atomicAdd(at + base + 0, norm * v.x);
        atomicAdd(at + base + 1, norm * v.y);
        atomicAdd(at + base + 2, norm * v.z);
        atomicAdd(at + base + 3, norm * v.w);
    }
}

// h = relu(agg + b1), in place
__global__ void gnn_relu_bias(float* __restrict__ agg, const float* __restrict__ b1, int total) {
    int i = blockIdx.x * blockDim.x + threadIdx.x;
    if (i < total) {
        float v = agg[i] + b1[i % F_HID];
        agg[i] = v > 0.0f ? v : 0.0f;
    }
}

// ---------------------------------------------------------------- conv2 GEMV: outc = h @ W2
__global__ __launch_bounds__(256) void gnn_gemv_w2(
    const float* __restrict__ h, const float* __restrict__ W2,
    float* __restrict__ outc, int n_nodes)
{
    int row = blockIdx.x * 8 + (threadIdx.x >> 5);
    int lane = threadIdx.x & 31;
    if (row >= n_nodes) return;
    const float* hr = h + (size_t)row * F_HID;
    float s = 0.0f;
#pragma unroll
    for (int i = 0; i < F_HID / 32; ++i) s += hr[lane + i * 32] * W2[lane + i * 32];
#pragma unroll
    for (int off = 16; off > 0; off >>= 1) s += __shfl_down(s, off);
    if (lane == 0) outc[row] = s;
}

// out2[i] = outc[i] * dinv[i]^2   (self-loop term of conv2)
__global__ void gnn_self_term_scalar(const float* __restrict__ outc,
                                     const float* __restrict__ dinv,
                                     float* __restrict__ out2, int n) {
    int i = blockIdx.x * blockDim.x + threadIdx.x;
    if (i < n) { float d = dinv[i]; out2[i] = outc[i] * d * d; }
}

__global__ void gnn_scatter_scalar(const int* __restrict__ src, const int* __restrict__ tgt,
                                   const float* __restrict__ dinv,
                                   const float* __restrict__ outc,
                                   float* __restrict__ out2, int n_edges) {
    int e = blockIdx.x * blockDim.x + threadIdx.x;
    if (e < n_edges) {
        int s = src[e], t = tgt[e];
        atomicAdd(&out2[t], dinv[s] * dinv[t] * outc[s]);
    }
}

// out[i] = (x[i,0]==1) ? out2[i] + b2 : 0
__global__ void gnn_final_mask(const float* __restrict__ x, const float* __restrict__ out2,
                               const float* __restrict__ b2, float* __restrict__ out, int n) {
    int i = blockIdx.x * blockDim.x + threadIdx.x;
    if (i < n) out[i] = (x[(size_t)i * F_IN] == 1.0f) ? (out2[i] + b2[0]) : 0.0f;
}

// ---------------------------------------------------------------- launcher
extern "C" void kernel_launch(void* const* d_in, const int* in_sizes, int n_in,
                              void* d_out, int out_size, void* d_ws, size_t ws_size,
                              hipStream_t stream) {
    const float* x  = (const float*)d_in[0];
    const int*   ei = (const int*)  d_in[1];
    const float* W1 = (const float*)d_in[2];
    const float* b1 = (const float*)d_in[3];
    const float* W2 = (const float*)d_in[4];
    const float* b2 = (const float*)d_in[5];
    float* out = (float*)d_out;

    const int n_nodes = in_sizes[0] / F_IN;
    const int n_edges = in_sizes[1] / 2;
    const int* src = ei;
    const int* tgt = ei + n_edges;

    // workspace carve-up (256B aligned)
    char* w = (char*)d_ws;
    auto take = [&](size_t bytes) -> char* {
        char* p = w;
        w += (bytes + 255) & ~(size_t)255;
        return p;
    };
    float*  deg  = (float*) take((size_t)n_nodes * 4);            // becomes dinv in place
    __bf16* w1t  = (__bf16*)take((size_t)F_IN * F_HID * 2);
    float*  h0   = (float*) take((size_t)n_nodes * F_HID * 4);
    float*  agg  = (float*) take((size_t)n_nodes * F_HID * 4);
    float*  outc = (float*) take((size_t)n_nodes * 4);
    float*  out2 = (float*) take((size_t)n_nodes * 4);
    (void)ws_size; (void)n_in; (void)out_size;

    const int T = 256;
    const int bn  = (n_nodes + T - 1) / T;
    const int be  = (n_edges + T - 1) / T;
    const int tot = n_nodes * F_HID;
    const int bt  = (tot + T - 1) / T;

    // 1) degrees: deg = 1 (self loop) + in-degree; dinv = rsqrt(deg)
    gnn_fill_f32<<<bn, T, 0, stream>>>(deg, 1.0f, n_nodes);
    gnn_deg_accum<<<be, T, 0, stream>>>(tgt, deg, n_edges);
    gnn_rsqrt_inplace<<<bn, T, 0, stream>>>(deg, n_nodes);   // deg -> dinv
    float* dinv = deg;

    // 2) pack W1 -> bf16 transposed
    gnn_pack_w1<<<(F_IN * F_HID + T - 1) / T, T, 0, stream>>>(W1, w1t);

    // 3) h0 = x @ W1 (bf16 WMMA, f32 accumulate, LDS-staged A tile)
    gnn_gemm1_wmma<<<(n_nodes + GEMM_M - 1) / GEMM_M, T, 0, stream>>>(x, w1t, h0, n_nodes);

    // 4) conv1 aggregation + bias + relu
    gnn_self_term<<<bt, T, 0, stream>>>(h0, dinv, agg, tot);
    gnn_scatter_rows<<<(n_edges + 7) / 8, T, 0, stream>>>(src, tgt, dinv, h0, agg, n_edges);
    gnn_relu_bias<<<bt, T, 0, stream>>>(agg, b1, tot);

    // 5) conv2: GEMV + aggregation
    gnn_gemv_w2<<<(n_nodes + 7) / 8, T, 0, stream>>>(agg, W2, outc, n_nodes);
    gnn_self_term_scalar<<<bn, T, 0, stream>>>(outc, dinv, out2, n_nodes);
    gnn_scatter_scalar<<<be, T, 0, stream>>>(src, tgt, dinv, outc, out2, n_edges);

    // 6) mask generators
    gnn_final_mask<<<bn, T, 0, stream>>>(x, out2, b2, out, n_nodes);
}
